// AdvancedAttention_8555574853649
// MI455X (gfx1250) — matrix operand order repaired and checked by reading, not compiled, not run on hardware
//
#include <hip/hip_runtime.h>
#include <hip/hip_bf16.h>
#include <stdint.h>

// ---- CDNA5 WMMA types (wave32) ----
typedef __attribute__((ext_vector_type(16))) __bf16 v16bf;
typedef __attribute__((ext_vector_type(8)))  float  v8f;

union Frag16 { v16bf v; unsigned int u[8]; };

__device__ __forceinline__ unsigned short f2bf(float f) {
    unsigned int u = __float_as_uint(f);
    u += 0x7FFFu + ((u >> 16) & 1u);        // round-to-nearest-even
    return (unsigned short)(u >> 16);
}
__device__ __forceinline__ float bf2f(unsigned short h) {
    return __uint_as_float(((unsigned int)h) << 16);
}

// ---- CDNA5 async global->LDS (ASYNCcnt-tracked, no VGPR bounce) ----
__device__ __forceinline__ unsigned int lds_addr32(const void* p) {
    // generic LDS pointer: aperture in [63:32], LDS offset in [31:0] (ISA 10.2)
    return (unsigned int)(size_t)p;
}
__device__ __forceinline__ void async_load_b128(unsigned int lds_off, const void* gaddr) {
    asm volatile("global_load_async_to_lds_b128 %0, %1, off"
                 :: "v"(lds_off), "v"(gaddr) : "memory");
}
__device__ __forceinline__ void wait_async0() {
    asm volatile("s_wait_asynccnt 0" ::: "memory");
}

// ---------------- f32 -> bf16 conversion ----------------
__global__ __launch_bounds__(256) void cvt_f32_bf16(const float* __restrict__ in,
                                                    unsigned short* __restrict__ out,
                                                    int n) {
    int i = blockIdx.x * 256 + threadIdx.x;
    if (i < n) out[i] = f2bf(in[i]);
}

// ---------------- WMMA GEMM: C[M,N] = A[M,K] * B[N,K]^T ----------------
// Block: 256 threads = 8 waves. Block tile 64(M) x 256(N), K-step 64,
// double-buffered async global->LDS staging. Wave tile 32x64 (2x4 WMMA).
__global__ __launch_bounds__(256) void gemm_bf16_wmma(
    const unsigned short* __restrict__ A,   // [M,K] bf16
    const unsigned short* __restrict__ B,   // [N,K] bf16
    unsigned short* __restrict__ Cb,        // bf16 out (if !out_f32)
    float* __restrict__ Cf,                 // f32 out  (if  out_f32)
    int M, int N, int K, int out_f32)
{
    __shared__ unsigned short als[2][64 * 64];     // 2 x  8 KB
    __shared__ unsigned short bls[2][256 * 64];    // 2 x 32 KB

    const int t    = threadIdx.x;
    const int lane = t & 31;
    const int wave = t >> 5;
    const int wm   = wave >> 2;          // 0..1  (M strip of 32)
    const int wn   = wave & 3;           // 0..3  (N strip of 64)
    const int M0   = blockIdx.y * 64;
    const int N0   = blockIdx.x * 256;
    const int l16  = lane & 15;
    const int lh   = lane >> 4;

    v8f acc[2][4];
    for (int i = 0; i < 2; ++i)
        for (int j = 0; j < 4; ++j)
            for (int r = 0; r < 8; ++r) acc[i][j][r] = 0.f;

    const int am = t >> 2, ako = (t & 3) * 16;   // A stage: 16 halfs/thread
    const int bn = t;                            // B stage: one 64-half row/thread

    const unsigned short* aRow = A + (size_t)(M0 + am) * K + ako;
    const unsigned short* bRow = B + (size_t)(N0 + bn) * K;

    auto issue_tile = [&](int buf, int kc) {
        unsigned int al = lds_addr32(&als[buf][am * 64 + ako]);
        const unsigned short* ag = aRow + kc;
        async_load_b128(al,      ag);
        async_load_b128(al + 16, ag + 8);
        unsigned int bl = lds_addr32(&bls[buf][bn * 64]);
        const unsigned short* bg = bRow + kc;
        #pragma unroll
        for (int u = 0; u < 8; ++u) async_load_b128(bl + u * 16, bg + u * 8);
    };

    issue_tile(0, 0);
    int buf = 0;
    for (int kc = 0; kc < K; kc += 64) {
        wait_async0();          // this wave's tile is in LDS
        __syncthreads();        // everyone's tile is in LDS; prev compute finished
        if (kc + 64 < K) issue_tile(buf ^ 1, kc + 64);   // overlap with compute

        const unsigned short* alsb = als[buf];
        const unsigned short* blsb = bls[buf];
        #pragma unroll
        for (int c = 0; c < 2; ++c) {            // two 32-deep WMMA K-chunks
            Frag16 af[2], bfr[4];
            #pragma unroll
            for (int i = 0; i < 2; ++i) {
                const unsigned short* rowp = alsb + (wm * 32 + i * 16 + l16) * 64 + c * 32;
                #pragma unroll
                for (int v = 0; v < 8; ++v) {
                    // A layout: M = lane%16; K = (v/4)*16 + (lane/16)*8 + (v%4)*2
                    int k = ((v >> 2) << 4) + (lh << 3) + ((v & 3) << 1);
                    af[i].u[v] = *(const unsigned int*)(rowp + k);
                }
            }
            #pragma unroll
            for (int j = 0; j < 4; ++j) {
                const unsigned short* rowp = blsb + (wn * 64 + j * 16 + l16) * 64 + c * 32;
                #pragma unroll
                for (int v = 0; v < 8; ++v) {
                    // B layout: N = lane%16; K = (v/4)*16 + (lane/16)*8 + (v%4)*2
                    int k = ((v >> 2) << 4) + (lh << 3) + ((v & 3) << 1);
                    bfr[j].u[v] = *(const unsigned int*)(rowp + k);
                }
            }
            #pragma unroll
            for (int i = 0; i < 2; ++i)
                #pragma unroll
                for (int j = 0; j < 4; ++j)
                    acc[i][j] = __builtin_amdgcn_wmma_f32_16x16x32_bf16(
                        false, af[i].v, false, bfr[j].v, (short)0, acc[i][j], false, false);
        }
        buf ^= 1;
    }

    // C/D layout: lane holds col N = lane%16; rows M = (lane/16)*8 + r
    for (int i = 0; i < 2; ++i)
        for (int j = 0; j < 4; ++j)
            for (int r = 0; r < 8; ++r) {
                int row = M0 + wm * 32 + i * 16 + lh * 8 + r;
                int col = N0 + wn * 64 + j * 16 + l16;
                float val = acc[i][j][r];
                if (out_f32) Cf[(size_t)row * N + col] = val;
                else         Cb[(size_t)row * N + col] = f2bf(val);
            }
}

// ---------------- RoPE (interleaved pairs) on bf16 [B,S,H,128] ----------------
__global__ __launch_bounds__(256) void rope_bf16(unsigned short* __restrict__ t,
                                                 const float* __restrict__ cs,
                                                 const float* __restrict__ sn,
                                                 int S, int H, int npairs) {
    int idx = blockIdx.x * 256 + threadIdx.x;
    if (idx >= npairs) return;
    int i    = idx & 63;          // pair index within head_dim/2
    int rest = idx >> 6;          // b*S*H flat
    int s    = (rest / H) % S;
    float c  = cs[s * 64 + i];
    float si = sn[s * 64 + i];
    unsigned short* p = t + (size_t)idx * 2;
    float tr = bf2f(p[0]);
    float ti = bf2f(p[1]);
    p[0] = f2bf(tr * c - ti * si);
    p[1] = f2bf(tr * si + ti * c);
}

// ---------------- Flash attention (causal, GQA) ----------------
// grid: (S/64, H=32, B). block: 128 threads = 4 waves; each wave owns 16 q rows.
// Double-buffered async K/V tile staging.
__global__ __launch_bounds__(128) void flash_attn_wmma(
    const unsigned short* __restrict__ Q,   // [B,S,32,128] bf16 (rope'd)
    const unsigned short* __restrict__ Km,  // [B,S,8,128]  bf16 (rope'd)
    const unsigned short* __restrict__ Vm,  // [B,S,8,128]  bf16
    unsigned short* __restrict__ O,         // [B,S,4096]   bf16
    int S)
{
    __shared__ unsigned short kls[2][32 * 128];   // K tiles [key][d]
    __shared__ unsigned short vls[2][32 * 128];   // V tiles [key][d]
    __shared__ unsigned short ps[4][16][32];      // per-wave P (probs) [q][key]

    const int t    = threadIdx.x;
    const int lane = t & 31;
    const int wave = t >> 5;
    const int l16  = lane & 15;
    const int lh   = lane >> 4;
    const int b    = blockIdx.z;
    const int h    = blockIdx.y;
    const int q0   = blockIdx.x * 64;
    const int kvh  = h >> 2;                   // N_REP = 4

    const float scale = 0.08838834764831845f;  // 1/sqrt(128)
    const float NEG   = -3.0e38f;

    const unsigned short* qbase = Q  + ((size_t)b * S) * 4096 + (size_t)h   * 128;
    const unsigned short* kbase = Km + ((size_t)b * S) * 1024 + (size_t)kvh * 128;
    const unsigned short* vbase = Vm + ((size_t)b * S) * 1024 + (size_t)kvh * 128;

    // Q fragments: 4 chunks of K(=d)=32, A-layout
    Frag16 qa[4];
    const int qrow = q0 + wave * 16 + l16;
    #pragma unroll
    for (int c = 0; c < 4; ++c)
        #pragma unroll
        for (int v = 0; v < 8; ++v) {
            int d = c * 32 + ((v >> 2) << 4) + (lh << 3) + ((v & 3) << 1);
            qa[c].u[v] = *(const unsigned int*)(qbase + (size_t)qrow * 4096 + d);
        }

    v8f oacc[8];
    for (int dt = 0; dt < 8; ++dt)
        for (int r = 0; r < 8; ++r) oacc[dt][r] = 0.f;
    float rmax[8], rsum[8];
    for (int r = 0; r < 8; ++r) { rmax[r] = NEG; rsum[r] = 0.f; }

    const int skey = t >> 2;            // staging: key row per thread
    const int sdo  = (t & 3) * 32;      // 32 halfs = 64B per thread
    const int nkt  = (q0 >> 5) + 2;     // causal: key tiles up to q0+63

    auto issue_kv = [&](int buf, int kt) {
        const unsigned short* kg = kbase + (size_t)(kt * 32 + skey) * 1024 + sdo;
        const unsigned short* vg = vbase + (size_t)(kt * 32 + skey) * 1024 + sdo;
        unsigned int ka = lds_addr32(&kls[buf][skey * 128 + sdo]);
        unsigned int va = lds_addr32(&vls[buf][skey * 128 + sdo]);
        #pragma unroll
        for (int u = 0; u < 4; ++u) {
            async_load_b128(ka + u * 16, kg + u * 8);
            async_load_b128(va + u * 16, vg + u * 8);
        }
    };

    issue_kv(0, 0);
    int buf = 0;
    for (int kt = 0; kt < nkt; ++kt) {
        const int k0 = kt * 32;
        wait_async0();
        __syncthreads();
        if (kt + 1 < nkt) issue_kv(buf ^ 1, kt + 1);

        const unsigned short* kb_ = kls[buf];
        const unsigned short* vb_ = vls[buf];

        // S = Q K^T  (two 16-key n-tiles, accumulate over 4 d-chunks)
        v8f s0, s1;
        for (int r = 0; r < 8; ++r) { s0[r] = 0.f; s1[r] = 0.f; }
        #pragma unroll
        for (int c = 0; c < 4; ++c) {
            Frag16 kb0, kb1;
            #pragma unroll
            for (int v = 0; v < 8; ++v) {
                int d = c * 32 + ((v >> 2) << 4) + (lh << 3) + ((v & 3) << 1);     // B: K-dim = d
                kb0.u[v] = *(const unsigned int*)(kb_ + l16 * 128 + d);
                kb1.u[v] = *(const unsigned int*)(kb_ + (16 + l16) * 128 + d);
            }
            s0 = __builtin_amdgcn_wmma_f32_16x16x32_bf16(false, qa[c].v, false, kb0.v, (short)0, s0, false, false);
            s1 = __builtin_amdgcn_wmma_f32_16x16x32_bf16(false, qa[c].v, false, kb1.v, (short)0, s1, false, false);
        }

        // online softmax (C layout: lane = column, rows = lh*8+r)
        #pragma unroll
        for (int r = 0; r < 8; ++r) {
            const int qr = q0 + wave * 16 + lh * 8 + r;
            float x0 = s0[r] * scale;
            float x1 = s1[r] * scale;
            if (k0 + l16 > qr)      x0 = NEG;
            if (k0 + 16 + l16 > qr) x1 = NEG;
            float m = fmaxf(x0, x1);
            m = fmaxf(m, __shfl_xor(m, 1, 32));
            m = fmaxf(m, __shfl_xor(m, 2, 32));
            m = fmaxf(m, __shfl_xor(m, 4, 32));
            m = fmaxf(m, __shfl_xor(m, 8, 32));
            float mnew  = fmaxf(rmax[r], m);
            float alpha = __expf(rmax[r] - mnew);
            float p0 = __expf(x0 - mnew);
            float p1 = __expf(x1 - mnew);
            float sm = p0 + p1;
            sm += __shfl_xor(sm, 1, 32);
            sm += __shfl_xor(sm, 2, 32);
            sm += __shfl_xor(sm, 4, 32);
            sm += __shfl_xor(sm, 8, 32);
            rsum[r] = rsum[r] * alpha + sm;
            rmax[r] = mnew;
            #pragma unroll
            for (int dt = 0; dt < 8; ++dt) oacc[dt][r] *= alpha;
            ps[wave][lh * 8 + r][l16]      = f2bf(p0);
            ps[wave][lh * 8 + r][16 + l16] = f2bf(p1);
        }

        // P back out of LDS in A-layout (per-wave region; DS is in-order per wave)
        Frag16 pa;
        #pragma unroll
        for (int v = 0; v < 8; ++v) {
            int k = ((v >> 2) << 4) + (lh << 3) + ((v & 3) << 1);
            pa.u[v] = *(const unsigned int*)(&ps[wave][l16][k]);
        }
        // O += P V  (8 d-tiles of 16)
        #pragma unroll
        for (int dt = 0; dt < 8; ++dt) {
            Frag16 vb;
            const int d = dt * 16 + l16;
            #pragma unroll
            for (int v = 0; v < 8; ++v) {
                int kk = ((v >> 2) << 4) + (lh << 3) + ((v & 3) << 1);             // B: K-dim = key
                unsigned int lo = vb_[kk * 128 + d];
                unsigned int hi = vb_[(kk + 1) * 128 + d];
                vb.u[v] = lo | (hi << 16);
            }
            oacc[dt] = __builtin_amdgcn_wmma_f32_16x16x32_bf16(false, pa.v, false, vb.v, (short)0, oacc[dt], false, false);
        }
        buf ^= 1;
    }

    // normalize and store [b, q, h*128 + d]
    #pragma unroll
    for (int dt = 0; dt < 8; ++dt)
        for (int r = 0; r < 8; ++r) {
            int row = q0 + wave * 16 + lh * 8 + r;
            int col = h * 128 + dt * 16 + l16;
            O[((size_t)b * S + row) * 4096 + col] = f2bf(oacc[dt][r] / rsum[r]);
        }
}

// ---------------- host-side orchestration ----------------
extern "C" void kernel_launch(void* const* d_in, const int* in_sizes, int n_in,
                              void* d_out, int out_size, void* d_ws, size_t ws_size,
                              hipStream_t stream) {
    (void)in_sizes; (void)n_in; (void)out_size; (void)ws_size;
    const float* x  = (const float*)d_in[0];
    // d_in[1] = mask (causal; applied analytically since start_pos == 0)
    const float* fc = (const float*)d_in[2];
    const float* fs = (const float*)d_in[3];
    const float* wq = (const float*)d_in[4];
    const float* wk = (const float*)d_in[5];
    const float* wv = (const float*)d_in[6];
    const float* wo = (const float*)d_in[7];
    float* out = (float*)d_out;

    const int B = 2, S = 2048, D = 4096, KV = 1024;
    const size_t MS = (size_t)B * S;             // 4096 rows

    // workspace carve (all bf16 = uint16)
    unsigned short* xb  = (unsigned short*)d_ws;
    unsigned short* wqb = xb  + MS * D;
    unsigned short* wkb = wqb + (size_t)D * D;
    unsigned short* wvb = wkb + (size_t)KV * D;
    unsigned short* wob = wvb + (size_t)KV * D;
    unsigned short* qb  = wob + (size_t)D * D;
    unsigned short* kb  = qb  + MS * D;
    unsigned short* vb  = kb  + MS * KV;
    unsigned short* ab  = vb  + MS * KV;

    cvt_f32_bf16<<<(int)((MS * D + 255) / 256), 256, 0, stream>>>(x,  xb,  (int)(MS * D));
    cvt_f32_bf16<<<(D * D + 255) / 256,          256, 0, stream>>>(wq, wqb, D * D);
    cvt_f32_bf16<<<(KV * D + 255) / 256,         256, 0, stream>>>(wk, wkb, KV * D);
    cvt_f32_bf16<<<(KV * D + 255) / 256,         256, 0, stream>>>(wv, wvb, KV * D);
    cvt_f32_bf16<<<(D * D + 255) / 256,          256, 0, stream>>>(wo, wob, D * D);

    // Q/K/V projections (block tile 64x256)
    gemm_bf16_wmma<<<dim3(D / 256,  (int)(MS / 64)), 256, 0, stream>>>(xb, wqb, qb, nullptr, (int)MS, D,  D, 0);
    gemm_bf16_wmma<<<dim3(KV / 256, (int)(MS / 64)), 256, 0, stream>>>(xb, wkb, kb, nullptr, (int)MS, KV, D, 0);
    gemm_bf16_wmma<<<dim3(KV / 256, (int)(MS / 64)), 256, 0, stream>>>(xb, wvb, vb, nullptr, (int)MS, KV, D, 0);

    // RoPE on Q and K
    int qpairs = B * S * 32 * 64;
    int kpairs = B * S * 8 * 64;
    rope_bf16<<<(qpairs + 255) / 256, 256, 0, stream>>>(qb, fc, fs, S, 32, qpairs);
    rope_bf16<<<(kpairs + 255) / 256, 256, 0, stream>>>(kb, fc, fs, S, 8,  kpairs);

    // causal GQA flash attention
    flash_attn_wmma<<<dim3(S / 64, 32, B), 128, 0, stream>>>(qb, kb, vb, ab, S);

    // output projection -> f32 d_out
    gemm_bf16_wmma<<<dim3(D / 256, (int)(MS / 64)), 256, 0, stream>>>(ab, wob, nullptr, out, (int)MS, D, D, 1);
}
